// MultiHeadedAttention_35622458753675
// MI455X (gfx1250) — matrix operand order repaired and checked by reading, not compiled, not run on hardware
//
#include <hip/hip_runtime.h>

typedef __attribute__((ext_vector_type(16))) _Float16 v16h;
typedef __attribute__((ext_vector_type(8)))  _Float16 v8h;
typedef __attribute__((ext_vector_type(8)))  float    v8f;
typedef __attribute__((ext_vector_type(4)))  unsigned int v4u;
typedef __attribute__((ext_vector_type(8)))  int v8i;
typedef __attribute__((ext_vector_type(4)))  int v4i;

constexpr int Bb    = 2;
constexpr int Tsq   = 2048;
constexpr int Dm    = 1024;
constexpr int Hn    = 16;
constexpr int DK    = 64;
constexpr int Mrows = Bb * Tsq;   // 4096

#if defined(__has_builtin)
#if __has_builtin(__builtin_amdgcn_tensor_load_to_lds)
#define ATTN_TDM 1
#endif
#endif

__device__ __forceinline__ v8f vzero8f() {
  v8f r;
#pragma unroll
  for (int i = 0; i < 8; ++i) r[i] = 0.0f;
  return r;
}

// Build a 16-half WMMA fragment from two 16-byte-aligned 8-half loads.
__device__ __forceinline__ v16h load_frag(const _Float16* p0, const _Float16* p1) {
  union { v16h v; v8h h[2]; } u;
  u.h[0] = *(const v8h*)p0;
  u.h[1] = *(const v8h*)p1;
  return u.v;
}

__device__ __forceinline__ v8f wmma16(v16h a, v16h b, v8f c) {
  // (neg_a, A, neg_b, B, c_mod, C, reuse_a, reuse_b)
  return __builtin_amdgcn_wmma_f32_16x16x32_f16(false, a, false, b, (short)0, c,
                                                false, false);
}

__device__ __forceinline__ unsigned lds_off(const void* p) {
  return (unsigned)(unsigned long long)p;   // LDS aperture: addr[31:0] = LDS byte offset
}

// ---- CDNA5 async copy: global -> LDS, tracked by ASYNCcnt (no VGPR round-trip).
__device__ __forceinline__ void async_ld_b128(unsigned ldsOff, const _Float16* g) {
  unsigned long long ga = (unsigned long long)g;
  asm volatile("global_load_async_to_lds_b128 %0, %1, off"
               :: "v"(ldsOff), "v"(ga) : "memory");
}
__device__ __forceinline__ void wait_async0() {
  asm volatile("s_wait_asynccnt 0x0" ::: "memory");
}

#ifdef ATTN_TDM
// ---- CDNA5 Tensor Data Mover: 2D tile (16-bit elements) global -> LDS.
// LDS destination padded by 8 halves (4 DWORDs) after every 64-half (32-DWORD) row:
// pad_interval code 4 (=32 DWORDs), pad_amount code 3 (=4 DWORDs).
__device__ __forceinline__ void tdm_load_2d_f16(unsigned ldsAddr, const void* gptr,
                                                unsigned tileD0, unsigned tileD1,
                                                unsigned stride0) {
  const unsigned long long ga = (unsigned long long)gptr;
  v4u g0;
  g0[0] = 1u;                                    // count=1, user descriptor
  g0[1] = ldsAddr;                               // lds_addr (bytes)
  g0[2] = (unsigned)ga;                          // global_addr[31:0]
  g0[3] = (unsigned)(ga >> 32) | (2u << 30);     // global_addr[56:32] | type=2
  v8i g1;
  g1[0] = (int)((1u << 16)        // data_size = 1 -> 2 bytes
              | (1u << 20)        // pad_enable
              | (4u << 22)        // pad_interval: every 32 DWORDs (one 64-half row)
              | (3u << 25));      // pad_amount: 4 DWORDs (8 halves)
  g1[1] = (int)(tileD0 << 16);                   // tensor_dim0[15:0]  @ bits 63:48
  g1[2] = (int)(tileD1 << 16);                   // tensor_dim1[15:0]  @ bits 95:80
  g1[3] = (int)(tileD0 << 16);                   // tile_dim0          @ bits 127:112
  g1[4] = (int)tileD1;                           // tile_dim1          @ bits 143:128
  g1[5] = (int)stride0;                          // tensor_dim0_stride[31:0]
  g1[6] = 0;
  g1[7] = 0;
  v4i gz4;
  gz4[0] = 0; gz4[1] = 0; gz4[2] = 0; gz4[3] = 0;   // groups 2/3 unused (2D tensor)
  v8i gz8;
#pragma unroll
  for (int i = 0; i < 8; ++i) gz8[i] = 0;
  // 6-arg form (clang-23 / therock-10.0 headers): (g0, g1, g2, g3, extra, cpol)
  __builtin_amdgcn_tensor_load_to_lds(g0, g1, gz4, gz4, gz8, 0);
}
#endif

// ---- Register tile for double-buffered GEMM staging (32 elements / thread).
template<typename TIn> struct RegTile;
template<> struct RegTile<float>    { float4 v[8]; };
template<> struct RegTile<_Float16> { v8h    v[4]; };

__device__ __forceinline__ void ldreg(const float* s, RegTile<float>& r) {
  const float4* s4 = (const float4*)s;
#pragma unroll
  for (int i = 0; i < 8; ++i) r.v[i] = s4[i];
}
__device__ __forceinline__ void ldreg(const _Float16* s, RegTile<_Float16>& r) {
  const v8h* s8 = (const v8h*)s;
#pragma unroll
  for (int i = 0; i < 4; ++i) r.v[i] = s8[i];
}
__device__ __forceinline__ void streg(const RegTile<float>& r, _Float16* d) {
#pragma unroll
  for (int c = 0; c < 4; ++c) {
    const float4 a = r.v[2 * c], b = r.v[2 * c + 1];
    v8h h;
    h[0] = (_Float16)a.x; h[1] = (_Float16)a.y; h[2] = (_Float16)a.z; h[3] = (_Float16)a.w;
    h[4] = (_Float16)b.x; h[5] = (_Float16)b.y; h[6] = (_Float16)b.z; h[7] = (_Float16)b.w;
    *(v8h*)(d + 8 * c) = h;
  }
}
__device__ __forceinline__ void streg(const RegTile<_Float16>& r, _Float16* d) {
#pragma unroll
  for (int c = 0; c < 4; ++c) *(v8h*)(d + 8 * c) = r.v[c];
}

// C = X @ W^T + bias.  M=4096, N=1024, K=1024 fixed.
// MODE 0: f16 output, head-split layout [B*H, T, 64], scaled by oscale.
// MODE 1: f32 output, row-major [M, N].
// Double-buffered: registers hold tile t+1 while WMMAs consume LDS tile t.
template<int MODE, typename TIn>
__global__ __launch_bounds__(256) void gemm_xwT(
    const TIn* __restrict__ X, const float* __restrict__ W,
    const float* __restrict__ bias, _Float16* __restrict__ outH,
    float* __restrict__ outF, float oscale)
{
  constexpr int BM = 128, BN = 128, BK = 64;
  constexpr int K = Dm;
  constexpr int LDK = BK + 8;              // 72 halves: rows stay 16B-aligned
  __shared__ _Float16 Ah[2][BM][LDK];      // X tile [m][k]
  __shared__ _Float16 Bh[2][BN][LDK];      // W tile [n][k] (rows of W = out cols)

  const int tid  = threadIdx.x;
  const int lane = tid & 31;
  const int wave = tid >> 5;
  const int wm   = wave & 3;    // wave M offset: 32*wm
  const int wn   = wave >> 2;   // wave N offset: 64*wn
  const int mb0  = blockIdx.y * BM;
  const int nb0  = blockIdx.x * BN;

  v8f acc[2][4];
#pragma unroll
  for (int i = 0; i < 2; ++i)
#pragma unroll
    for (int j = 0; j < 4; ++j) acc[i][j] = vzero8f();

  const int lrow = tid >> 1;          // 0..127
  const int lcol = (tid & 1) * 32;    // 0 or 32
  const int kb0h = (lane < 16) ? 0 : 8;

  RegTile<TIn>  ra;
  RegTile<float> rb;
  ldreg(X + (size_t)(mb0 + lrow) * K + lcol, ra);
  ldreg(W + (size_t)(nb0 + lrow) * K + lcol, rb);

  int buf = 0;
  for (int kb = 0; kb < K; kb += BK, buf ^= 1) {
    streg(ra, &Ah[buf][lrow][lcol]);
    streg(rb, &Bh[buf][lrow][lcol]);
    __syncthreads();
    if (kb + BK < K) {
      ldreg(X + (size_t)(mb0 + lrow) * K + kb + BK + lcol, ra);
      ldreg(W + (size_t)(nb0 + lrow) * K + kb + BK + lcol, rb);
      if (kb + 2 * BK < K) {
        __builtin_prefetch(X + (size_t)(mb0 + lrow) * K + kb + 2 * BK + lcol, 0, 0);
        __builtin_prefetch(W + (size_t)(nb0 + lrow) * K + kb + 2 * BK + lcol, 0, 0);
      }
    }
#pragma unroll
    for (int st = 0; st < 2; ++st) {
      v16h a[2], b[4];
#pragma unroll
      for (int mf = 0; mf < 2; ++mf) {
        const _Float16* p = &Ah[buf][wm * 32 + mf * 16 + (lane & 15)][st * 32 + kb0h];
        a[mf] = load_frag(p, p + 16);
      }
#pragma unroll
      for (int nf = 0; nf < 4; ++nf) {
        const _Float16* p = &Bh[buf][wn * 64 + nf * 16 + (lane & 15)][st * 32 + (lane >> 4) * 8];
        b[nf] = load_frag(p, p + 16);
      }
#pragma unroll
      for (int mf = 0; mf < 2; ++mf)
#pragma unroll
        for (int nf = 0; nf < 4; ++nf)
          acc[mf][nf] = wmma16(a[mf], b[nf], acc[mf][nf]);
    }
    // Single barrier per iteration: next iter stores into buf^1, which no wave
    // can still be reading (it last read buf^1 before the previous barrier).
  }
  __syncthreads();

#pragma unroll
  for (int mf = 0; mf < 2; ++mf)
#pragma unroll
    for (int nf = 0; nf < 4; ++nf)
#pragma unroll
      for (int vr = 0; vr < 8; ++vr) {
        const int row = mb0 + wm * 32 + mf * 16 + vr + ((lane >> 4) << 3);
        const int col = nb0 + wn * 64 + nf * 16 + (lane & 15);
        const float v = acc[mf][nf][vr] + bias[col];
        if constexpr (MODE == 0) {
          const int bi = row >> 11, t = row & (Tsq - 1);
          const int h  = col >> 6,  dk = col & 63;
          outH[((size_t)(bi * Hn + h) * Tsq + t) * DK + dk] = (_Float16)(v * oscale);
        } else {
          outF[(size_t)row * Dm + col] = v;
        }
      }
}

// Causal flash attention. Q/K/V: f16 [B*H, T, 64] (Q pre-scaled by 1/sqrt(64)).
// 64-query block (4 waves x 16 queries), 64-key tiles.
// K tile staged via TDM (tensor_load_to_lds with HW row padding) when available,
// else via CDNA5 async global->LDS copies; V staged transposed manually.
__global__ __launch_bounds__(128) void attn_fwd(
    const _Float16* __restrict__ Qh, const _Float16* __restrict__ Kh,
    const _Float16* __restrict__ Vh, _Float16* __restrict__ Cx)
{
  constexpr int LDB = 72;                 // padded row: 144B, 16B-aligned
  __shared__ _Float16 Kt[64][LDB];        // [key][dk]   (B-frags for Q*K^T)
  __shared__ _Float16 Vt[64][LDB];        // [dk][key]   (B-frags for P*V)
  __shared__ _Float16 Pb[4][16][LDB];     // per-wave P staging (16 x 64)

  const int tid   = threadIdx.x;
  const int lane  = tid & 31;
  const int wave  = tid >> 5;
  const int bh    = blockIdx.y;              // 0..B*H-1
  const int qb0   = blockIdx.x * 64;         // block query base
  const int qrow0 = qb0 + wave * 16;         // wave query tile base
  const int kb0h  = (lane < 16) ? 0 : 8;
  const size_t headBase = (size_t)bh * Tsq * DK;

  // Q A-fragments for the two k=32 steps over DK=64 (kept in registers).
  v16h aq[2];
  {
    const _Float16* qp = Qh + headBase + (size_t)(qrow0 + (lane & 15)) * DK;
    aq[0] = load_frag(qp + kb0h,      qp + kb0h + 16);
    aq[1] = load_frag(qp + 32 + kb0h, qp + 32 + kb0h + 16);
  }

  v8f acc[4];
#pragma unroll
  for (int nt = 0; nt < 4; ++nt) acc[nt] = vzero8f();
  float mrow[8], lsum[8];
#pragma unroll
  for (int vr = 0; vr < 8; ++vr) { mrow[vr] = -1e30f; lsum[vr] = 0.0f; }

  const int nkt = qb0 / 64 + 1;   // 64-key tiles needed by this block (causal)
  for (int kt = 0; kt < nkt; ++kt) {
    const int kbase = kt * 64;
    {
#ifdef ATTN_TDM
      // K: one TDM descriptor moves the whole 64x64 f16 tile, HW-padding each
      // 64-half row by 8 halves to match Kt's LDS stride. Wave 0 issues it.
      if (wave == 0)
        tdm_load_2d_f16(lds_off(&Kt[0][0]),
                        Kh + headBase + (size_t)kbase * DK,
                        64u, 64u, (unsigned)DK);
#else
      // K: async bulk copy, 64 rows x 128B; each thread issues 4 b128 copies.
      const int r0  = tid >> 3;         // 0..15
      const int seg = tid & 7;          // 16B segment in row
#pragma unroll
      for (int rr = 0; rr < 4; ++rr) {
        const int r = r0 + rr * 16;
        async_ld_b128(lds_off(&Kt[r][seg * 8]),
                      Kh + headBase + (size_t)(kbase + r) * DK + seg * 8);
      }
#endif
      // V: load 32 contiguous halves, scatter transposed into Vt[dk][key].
      const int vrow = tid >> 1;           // key 0..63
      const int cb   = (tid & 1) * 32;     // dk base
      const _Float16* vs = Vh + headBase + (size_t)(kbase + vrow) * DK + cb;
      const v8h v0 = ((const v8h*)vs)[0];
      const v8h v1 = ((const v8h*)vs)[1];
      const v8h v2 = ((const v8h*)vs)[2];
      const v8h v3 = ((const v8h*)vs)[3];
#pragma unroll
      for (int j = 0; j < 8; ++j) Vt[cb + j][vrow]      = v0[j];
#pragma unroll
      for (int j = 0; j < 8; ++j) Vt[cb + 8 + j][vrow]  = v1[j];
#pragma unroll
      for (int j = 0; j < 8; ++j) Vt[cb + 16 + j][vrow] = v2[j];
#pragma unroll
      for (int j = 0; j < 8; ++j) Vt[cb + 24 + j][vrow] = v3[j];
#ifdef ATTN_TDM
      __builtin_amdgcn_s_wait_tensorcnt(0);   // no-op for waves 1..3
#else
      wait_async0();
#endif
    }
    __syncthreads();

    if (kbase <= qrow0 + 15) {
      // S = Q K^T for 4 key sub-tiles of 16.
      v8f s[4];
#pragma unroll
      for (int nt = 0; nt < 4; ++nt) {
        v8f c = vzero8f();
#pragma unroll
        for (int st = 0; st < 2; ++st) {
          const _Float16* bp = &Kt[nt * 16 + (lane & 15)][st * 32 + (lane >> 4) * 8];
          c = wmma16(aq[st], load_frag(bp, bp + 16), c);
        }
        s[nt] = c;
      }
      // Causal mask (only a tile overlapping the diagonal can violate).
      if (kbase + 63 > qrow0) {
#pragma unroll
        for (int nt = 0; nt < 4; ++nt)
#pragma unroll
          for (int vr = 0; vr < 8; ++vr) {
            const int kj = kbase + nt * 16 + (lane & 15);
            const int qi = qrow0 + vr + ((lane >> 4) << 3);
            if (kj > qi) s[nt][vr] = -1e30f;
          }
      }
      // Online softmax over the 64 new scores per row.
#pragma unroll
      for (int vr = 0; vr < 8; ++vr) {
        float mx = fmaxf(fmaxf(s[0][vr], s[1][vr]), fmaxf(s[2][vr], s[3][vr]));
#pragma unroll
        for (int off = 8; off; off >>= 1) mx = fmaxf(mx, __shfl_xor(mx, off, 16));
        const float mnew = fmaxf(mrow[vr], mx);
        const float corr = __expf(mrow[vr] - mnew);
        mrow[vr] = mnew;
        float ps = 0.0f;
#pragma unroll
        for (int nt = 0; nt < 4; ++nt) {
          const float p = __expf(s[nt][vr] - mnew);
          s[nt][vr] = p;
          ps += p;
        }
#pragma unroll
        for (int off = 8; off; off >>= 1) ps += __shfl_xor(ps, off, 16);
        lsum[vr] = lsum[vr] * corr + ps;
#pragma unroll
        for (int nt = 0; nt < 4; ++nt) acc[nt][vr] *= corr;
      }
      // P: C-layout -> per-wave LDS -> A-layout (same-wave LDS ops in-order).
#pragma unroll
      for (int nt = 0; nt < 4; ++nt)
#pragma unroll
        for (int vr = 0; vr < 8; ++vr)
          Pb[wave][vr + ((lane >> 4) << 3)][nt * 16 + (lane & 15)] = (_Float16)s[nt][vr];

#pragma unroll
      for (int st = 0; st < 2; ++st) {
        const _Float16* pp = &Pb[wave][lane & 15][st * 32 + kb0h];
        const v16h pa = load_frag(pp, pp + 16);
#pragma unroll
        for (int nt = 0; nt < 4; ++nt) {
          const _Float16* vp = &Vt[nt * 16 + (lane & 15)][st * 32 + (lane >> 4) * 8];
          acc[nt] = wmma16(pa, load_frag(vp, vp + 16), acc[nt]);
        }
      }
    }
    __syncthreads();
  }

  // Normalize and write ctx as f16 [B*T, D].
  const int bi = bh >> 4, h = bh & 15;
#pragma unroll
  for (int vr = 0; vr < 8; ++vr) {
    const float inv = 1.0f / lsum[vr];
    const int qi = qrow0 + vr + ((lane >> 4) << 3);
    const size_t ro = ((size_t)(bi * Tsq + qi)) * Dm + h * DK;
#pragma unroll
    for (int nt = 0; nt < 4; ++nt)
      Cx[ro + nt * 16 + (lane & 15)] = (_Float16)(acc[nt][vr] * inv);
  }
}

extern "C" void kernel_launch(void* const* d_in, const int* in_sizes, int n_in,
                              void* d_out, int out_size, void* d_ws, size_t ws_size,
                              hipStream_t stream) {
  const float* q  = (const float*)d_in[0];
  const float* k  = (const float*)d_in[1];
  const float* v  = (const float*)d_in[2];
  // d_in[3] = attn_mask (causal triu; applied analytically in attn_fwd)
  const float* Wq = (const float*)d_in[4];
  const float* bq = (const float*)d_in[5];
  const float* Wk = (const float*)d_in[6];
  const float* bk = (const float*)d_in[7];
  const float* Wv = (const float*)d_in[8];
  const float* bv = (const float*)d_in[9];
  const float* Wo = (const float*)d_in[10];
  const float* bo = (const float*)d_in[11];
  float* out = (float*)d_out;

  const size_t elems = (size_t)Mrows * Dm;       // 4,194,304 halves each
  _Float16* Qh = (_Float16*)d_ws;
  _Float16* Kh = Qh + elems;
  _Float16* Vh = Kh + elems;
  _Float16* Cx = Vh + elems;                     // total ws use: 32 MB

  const dim3 gblk(256);
  const dim3 ggrid(Dm / 128, Mrows / 128);       // (8, 32)

  // QKV projections (1/sqrt(DK) folded into Q).
  gemm_xwT<0, float><<<ggrid, gblk, 0, stream>>>(q, Wq, bq, Qh, nullptr, 0.125f);
  gemm_xwT<0, float><<<ggrid, gblk, 0, stream>>>(k, Wk, bk, Kh, nullptr, 1.0f);
  gemm_xwT<0, float><<<ggrid, gblk, 0, stream>>>(v, Wv, bv, Vh, nullptr, 1.0f);

  // Causal flash attention: grid (T/64 query blocks, B*H heads).
  const dim3 agrid(Tsq / 64, Bb * Hn);
  attn_fwd<<<agrid, 128, 0, stream>>>(Qh, Kh, Vh, Cx);

  // Output projection to f32.
  gemm_xwT<1, _Float16><<<ggrid, gblk, 0, stream>>>(Cx, Wo, bo, nullptr, out, 1.0f);
}
